// AFTGeneral_67379446940302
// MI455X (gfx1250) — matrix operand order repaired and checked by reading, not compiled, not run on hardware
//
#include <hip/hip_runtime.h>
#include <hip/hip_bf16.h>

typedef __attribute__((ext_vector_type(16))) _Float16 v16h;
typedef __attribute__((ext_vector_type(8)))  _Float16 v8h;
typedef __attribute__((ext_vector_type(8)))  float    v8f;

#define NSEQ 1024
#define DIMD 128

// ---------- WMMA helpers (CDNA5 wave32, 16x16x32 f16 -> f32) ----------

__device__ inline v8f wmma_f16(v16h a, v16h b, v8f c) {
  // 8 args: (neg_a, A, neg_b, B, c_mod, C, reuse_a, reuse_b)
  return __builtin_amdgcn_wmma_f32_16x16x32_f16(false, a, false, b, (short)0, c,
                                                false, false);
}

__device__ inline int wave_lane() { return threadIdx.x & 31; }

// A fragment: 16x32 f16 tile from row-major f16 src (leading dim ld halves).
// Lane L: row = L&15; halves 0..15 = K[off..off+16),
// off = 16*(L>>4). Both runs contiguous & 16B aligned -> two global_load_b128.
__device__ inline v16h load_a_frag(const _Float16* __restrict__ src, int ld,
                                   int row0, int k0) {
  const int lane = wave_lane();
  const int row  = row0 + (lane & 15);
  const int off  = (lane >> 4) * 16;
  const _Float16* p = src + (size_t)row * ld + k0 + off;
  v8h lo = *(const v8h*)p;
  v8h hi = *(const v8h*)(p + 8);
  return __builtin_shufflevector(lo, hi, 0, 1, 2, 3, 4, 5, 6, 7,
                                 8, 9, 10, 11, 12, 13, 14, 15);
}

// ---- Packed-B layout ----
// A 32x16 B fragment (k-tile k0, col-tile col0) is stored as 32 lanes x 16
// contiguous halves (32B/lane). Element (kk, col) of a KxN matrix lives at:
//   frag = (kk/32)*(N/16) + col/16 ; lane = (col&15) + 16*((kk>>4)&1) ; h = kk&15
__device__ __host__ inline size_t b_packed_index(int kk, int col, int N) {
  const int frag = (kk >> 5) * (N >> 4) + (col >> 4);
  const int lane = (col & 15) + (((kk >> 4) & 1) << 4);
  return (size_t)frag * 512 + (size_t)lane * 16 + (kk & 15);
}

// One aligned 32B read per lane -> two global_load_b128.
__device__ inline v16h load_b_frag_packed(const _Float16* __restrict__ src, int N,
                                          int k0, int col0) {
  const int frag = (k0 >> 5) * (N >> 4) + (col0 >> 4);
  const v16h* p = (const v16h*)(src + (size_t)frag * 512);
  return p[wave_lane()];
}

// f32 C/D layout: VGPR r -> (M = 8*(lane>>4) + r, N = lane&15)
__device__ inline int cd_index(int row0, int col0, int r, int ldc) {
  const int lane = wave_lane();
  return (row0 + ((lane >> 4) * 8) + r) * ldc + col0 + (lane & 15);
}

// ---------- elementwise / setup kernels ----------

__global__ void f32_to_f16_kernel(const float* __restrict__ src,
                                  _Float16* __restrict__ dst, int n) {
  int i = blockIdx.x * blockDim.x + threadIdx.x;
  if (i < n) dst[i] = (_Float16)src[i];
}

// Convert an f32 KxN weight directly into packed-B f16 layout.
__global__ void pack_b_f16_kernel(const float* __restrict__ src,
                                  _Float16* __restrict__ dst, int K, int N) {
  int i = blockIdx.x * blockDim.x + threadIdx.x;
  if (i >= K * N) return;
  int kk = i / N, col = i % N;
  dst[b_packed_index(kk, col, N)] = (_Float16)src[i];
}

// column max of k over the m axis (exact softmax stabilization; cancels in num/den)
__global__ void colmax_kernel(const float* __restrict__ k, float* __restrict__ cmax) {
  int d = threadIdx.x;  // 128 threads
  float m = -3.0e38f;
#pragma unroll 4
  for (int i = 0; i < NSEQ; ++i) m = fmaxf(m, k[i * DIMD + d]);
  cmax[d] = m;
}

// eK = exp(k - colmax), eKv = eK * v — written straight into packed-B layout.
__global__ void expk_kernel(const float* __restrict__ k, const float* __restrict__ v,
                            const float* __restrict__ cmax,
                            _Float16* __restrict__ eKh, _Float16* __restrict__ eKvh) {
  int i = blockIdx.x * blockDim.x + threadIdx.x;  // m*DIMD + d
  int m = i >> 7, d = i & (DIMD - 1);
  float e = __expf(k[i] - cmax[d]);
  size_t pi = b_packed_index(m, d, DIMD);
  eKh[pi]  = (_Float16)e;
  eKvh[pi] = (_Float16)(e * v[i]);
}

// E[n,m] = exp(pu[n] * pv[m]) — A operand, row-major.
__global__ void bias_exp_kernel(const float* __restrict__ pu,
                                const float* __restrict__ pv,
                                _Float16* __restrict__ Eh) {
  int i = blockIdx.x * blockDim.x + threadIdx.x;  // n*NSEQ + m
  int n = i >> 10, m = i & (NSEQ - 1);
  Eh[i] = (_Float16)__expf(pu[n] * pv[m]);
}

// ---------- WMMA GEMM kernels (8 waves/block, one tile strip per wave) ----------

// q = sigmoid(x@Wq), k = x@Wk, v = x@Wv — A (x) fragment shared across 3 accumulators
__global__ __launch_bounds__(256) void qkv_kernel(
    const _Float16* __restrict__ xh, const _Float16* __restrict__ wq,
    const _Float16* __restrict__ wk, const _Float16* __restrict__ wv,
    float* __restrict__ q, float* __restrict__ k, float* __restrict__ v) {
  const int tile = blockIdx.x * 8 + (threadIdx.x >> 5);  // 512 tiles (64 x 8)
  const int row0 = (tile >> 3) * 16;
  const int col0 = (tile & 7) * 16;
  v8f cq = {}, ck = {}, cv = {};
#pragma unroll
  for (int k0 = 0; k0 < DIMD; k0 += 32) {
    v16h a  = load_a_frag(xh, DIMD, row0, k0);
    v16h bq = load_b_frag_packed(wq, DIMD, k0, col0);
    v16h bk = load_b_frag_packed(wk, DIMD, k0, col0);
    v16h bv = load_b_frag_packed(wv, DIMD, k0, col0);
    cq = wmma_f16(a, bq, cq);
    ck = wmma_f16(a, bk, ck);
    cv = wmma_f16(a, bv, cv);
  }
#pragma unroll
  for (int r = 0; r < 8; ++r) {
    int idx = cd_index(row0, col0, r, DIMD);
    q[idx] = 1.0f / (1.0f + __expf(-cq[r]));
    k[idx] = ck[r];
    v[idx] = cv[r];
  }
}

// num = E@eKv, den = E@eK. Each wave: 16x32 strip (2 col tiles x {num,den}),
// E fragment shared by all 4 WMMAs. Epilogue: out = f16(q * num / den).
__global__ __launch_bounds__(256) void aft_core_kernel(
    const _Float16* __restrict__ Eh, const _Float16* __restrict__ eKh,
    const _Float16* __restrict__ eKvh, const float* __restrict__ q,
    _Float16* __restrict__ outh) {
  const int tile = blockIdx.x * 8 + (threadIdx.x >> 5);  // 256 strips (64 x 4)
  const int row0 = (tile >> 2) * 16;
  const int col0 = (tile & 3) * 32;
  v8f n0 = {}, d0 = {}, n1 = {}, d1 = {};
  for (int k0 = 0; k0 < NSEQ; k0 += 32) {
    v16h a   = load_a_frag(Eh, NSEQ, row0, k0);
    v16h bn0 = load_b_frag_packed(eKvh, DIMD, k0, col0);
    v16h bd0 = load_b_frag_packed(eKh,  DIMD, k0, col0);
    v16h bn1 = load_b_frag_packed(eKvh, DIMD, k0, col0 + 16);
    v16h bd1 = load_b_frag_packed(eKh,  DIMD, k0, col0 + 16);
    n0 = wmma_f16(a, bn0, n0);
    d0 = wmma_f16(a, bd0, d0);
    n1 = wmma_f16(a, bn1, n1);
    d1 = wmma_f16(a, bd1, d1);
  }
#pragma unroll
  for (int r = 0; r < 8; ++r) {
    int i0 = cd_index(row0, col0, r, DIMD);
    int i1 = cd_index(row0, col0 + 16, r, DIMD);
    outh[i0] = (_Float16)(q[i0] * n0[r] / d0[r]);
    outh[i1] = (_Float16)(q[i1] * n1[r] / d1[r]);
  }
}

// y = out @ Wo + bo
__global__ __launch_bounds__(256) void proj_kernel(
    const _Float16* __restrict__ outh, const _Float16* __restrict__ woh,
    const float* __restrict__ bo, float* __restrict__ y) {
  const int tile = blockIdx.x * 8 + (threadIdx.x >> 5);  // 512 tiles
  const int row0 = (tile >> 3) * 16;
  const int col0 = (tile & 7) * 16;
  v8f c = {};
#pragma unroll
  for (int k0 = 0; k0 < DIMD; k0 += 32) {
    v16h a = load_a_frag(outh, DIMD, row0, k0);
    v16h b = load_b_frag_packed(woh, DIMD, k0, col0);
    c = wmma_f16(a, b, c);
  }
  const float bias = bo[col0 + (wave_lane() & 15)];
#pragma unroll
  for (int r = 0; r < 8; ++r) {
    int idx = cd_index(row0, col0, r, DIMD);
    y[idx] = c[r] + bias;
  }
}

// ---------- host launcher ----------

extern "C" void kernel_launch(void* const* d_in, const int* in_sizes, int n_in,
                              void* d_out, int out_size, void* d_ws, size_t ws_size,
                              hipStream_t stream) {
  const float* x  = (const float*)d_in[0];   // [1,1024,128]
  const float* Wq = (const float*)d_in[1];   // [128,128]
  const float* Wk = (const float*)d_in[2];
  const float* Wv = (const float*)d_in[3];
  const float* Wo = (const float*)d_in[4];
  const float* bo = (const float*)d_in[5];   // [128]
  const float* pu = (const float*)d_in[6];   // [1024]
  const float* pv = (const float*)d_in[7];   // [1024]
  float* y = (float*)d_out;                  // [1,1024,128]

  char* ws = (char*)d_ws;
  size_t off = 0;
  auto alloc = [&](size_t bytes) -> void* {
    void* p = ws + off;
    off += (bytes + 255) & ~(size_t)255;
    return p;
  };
  const int NX = NSEQ * DIMD;     // 131072
  const int NW = DIMD * DIMD;     // 16384
  _Float16* xh   = (_Float16*)alloc(NX * 2);            // row-major A
  _Float16* wqh  = (_Float16*)alloc(NW * 2);            // packed B
  _Float16* wkh  = (_Float16*)alloc(NW * 2);            // packed B
  _Float16* wvh  = (_Float16*)alloc(NW * 2);            // packed B
  _Float16* woh  = (_Float16*)alloc(NW * 2);            // packed B
  float*    qf   = (float*)alloc(NX * 4);
  float*    kf   = (float*)alloc(NX * 4);
  float*    vf   = (float*)alloc(NX * 4);
  float*    cmax = (float*)alloc(DIMD * 4);
  _Float16* eKh  = (_Float16*)alloc(NX * 2);            // packed B
  _Float16* eKvh = (_Float16*)alloc(NX * 2);            // packed B
  _Float16* Eh   = (_Float16*)alloc((size_t)NSEQ * NSEQ * 2);  // row-major A
  _Float16* outh = (_Float16*)alloc(NX * 2);            // row-major A

  // operand prep
  f32_to_f16_kernel<<<(NX + 255) / 256, 256, 0, stream>>>(x, xh, NX);
  pack_b_f16_kernel<<<(NW + 255) / 256, 256, 0, stream>>>(Wq, wqh, DIMD, DIMD);
  pack_b_f16_kernel<<<(NW + 255) / 256, 256, 0, stream>>>(Wk, wkh, DIMD, DIMD);
  pack_b_f16_kernel<<<(NW + 255) / 256, 256, 0, stream>>>(Wv, wvh, DIMD, DIMD);
  pack_b_f16_kernel<<<(NW + 255) / 256, 256, 0, stream>>>(Wo, woh, DIMD, DIMD);

  qkv_kernel<<<64, 256, 0, stream>>>(xh, wqh, wkh, wvh, qf, kf, vf);
  colmax_kernel<<<1, DIMD, 0, stream>>>(kf, cmax);
  expk_kernel<<<NX / 256, 256, 0, stream>>>(kf, vf, cmax, eKh, eKvh);
  bias_exp_kernel<<<(NSEQ * NSEQ) / 256, 256, 0, stream>>>(pu, pv, Eh);
  aft_core_kernel<<<32, 256, 0, stream>>>(Eh, eKh, eKvh, qf, outh);
  proj_kernel<<<64, 256, 0, stream>>>(outh, woh, bo, y);
  (void)in_sizes; (void)n_in; (void)out_size; (void)ws_size;
}
